// HGDN_32169305047212
// MI455X (gfx1250) — compile-verified, matrix operand order checked by reading, not run on hardware
//
#include <hip/hip_runtime.h>
#include <hip/hip_bf16.h>

// ---------------------------------------------------------------------------
// Problem constants (fixed by the reference: B=64, N=1024, L=D=512, T=2, P=512)
// ---------------------------------------------------------------------------
#define CB   64
#define CN   1024
#define CL   512
#define CD   512
#define CT   2
#define CP   512
#define CK   30
#define ROWS_PER_T (CB * CP)      // 32768
#define HX_ROWS (CT * ROWS_PER_T) // 65536

typedef __attribute__((ext_vector_type(16))) __bf16 v16bf;
typedef __attribute__((ext_vector_type(8)))  __bf16 v8bf;
typedef __attribute__((ext_vector_type(4)))  __bf16 v4bf;
typedef __attribute__((ext_vector_type(8)))  float  v8f;

#define MODE_HX  0
#define MODE_HV  1
#define MODE_SIM 2

// ---------------------------------------------------------------------------
// bf16 WMMA GEMM:  C[t] = A[t] (MxK) * Bt[t]^T (+bias), Bt stored [N][K].
// Block tile 128x128x32, double-buffered LDS, software-pipelined:
//   loads(s+1) -> WMMA(s) -> ds_store(s+1, alt buf) -> one barrier.
// 8 waves (4x2); each wave owns 32x64 = 2 A-frags x 4 B-frags -> 8
// v_wmma_f32_16x16x32_bf16 per K-step per wave. All global->LDS staging is
// 16-byte chunks; all LDS->fragment loads are ds_load_b128 (two/fragment).
// ---------------------------------------------------------------------------
template <int MODE>
__global__ __launch_bounds__(256) void gemm_bf16_wmma(
    const __bf16* __restrict__ A, const __bf16* __restrict__ Bt,
    const float* __restrict__ bias, float* __restrict__ C,
    int M, int Ncol, int Kdim)
{
    constexpr int BM = 128, BN = 128, KP = 40;  // KP: 80B pitch (16B multiple)
    __shared__ __align__(16) __bf16 As[2][BM][KP];
    __shared__ __align__(16) __bf16 Bs[2][BN][KP];

    const int t      = blockIdx.y;
    const int ntiles = Ncol / BN;
    const int bm     = (blockIdx.x / ntiles) * BM;
    const int bn     = (blockIdx.x % ntiles) * BN;
    const int tid    = threadIdx.x;
    const int lane   = tid & 31;
    const int wave   = tid >> 5;
    const int wr     = (wave >> 1) * 32;    // wave row offset (4 row-groups)
    const int wc     = (wave & 1) * 64;     // wave col offset (2 col-groups)

    // ---- hoisted staging pointers (constant across k-loop) ----------------
    const int row0 = tid >> 2;              // rows row0 and row0+64
    const int koff = (tid & 3) * 8;         // bf16 elements (16B chunks)
    const __bf16* arp[2];
    #pragma unroll
    for (int i = 0; i < 2; ++i) {
        int n = bm + row0 + i * 64;
        size_t g;
        if (MODE == MODE_HX)      g = (size_t)((n >> 9) * CN + t * CP + (n & (CP - 1)));
        else if (MODE == MODE_HV) g = (size_t)(t * CP + n);
        else                      g = (size_t)n;
        arp[i] = A + g * (size_t)Kdim + koff;
    }
    const __bf16* brp[2];
    #pragma unroll
    for (int i = 0; i < 2; ++i)
        brp[i] = Bt + ((size_t)t * Ncol + (bn + row0 + i * 64)) * (size_t)Kdim + koff;

    v8f acc[2][4] = {};

    // ---- prologue: stage K-step 0 into buffer 0 ---------------------------
    {
        uint4 a0 = *(const uint4*)(arp[0]);
        uint4 a1 = *(const uint4*)(arp[1]);
        uint4 b0 = *(const uint4*)(brp[0]);
        uint4 b1 = *(const uint4*)(brp[1]);
        *(uint4*)&As[0][row0][koff]      = a0;
        *(uint4*)&As[0][row0 + 64][koff] = a1;
        *(uint4*)&Bs[0][row0][koff]      = b0;
        *(uint4*)&Bs[0][row0 + 64][koff] = b1;
    }
    __syncthreads();

    const int nsteps = Kdim / 32;
    const int ml  = lane & 15;
    const int khi = (lane >> 4) << 3;       // lanes 0-15 -> K base 0; 16-31 -> 8

    for (int s = 0; s < nsteps; ++s) {
        const int cur = s & 1;
        // ---- issue global loads for step s+1 (overlap with WMMA below) ----
        uint4 a0, a1, b0, b1;
        const bool more = (s + 1) < nsteps;
        if (more) {
            int k0 = (s + 1) * 32;
            a0 = *(const uint4*)(arp[0] + k0);
            a1 = *(const uint4*)(arp[1] + k0);
            b0 = *(const uint4*)(brp[0] + k0);
            b1 = *(const uint4*)(brp[1] + k0);
        }

        // ---- fragments: two ds_load_b128 each (ISA 7.12.2 wave32 layout) --
        v16bf af[2], bfr[4];
        #pragma unroll
        for (int i = 0; i < 2; ++i) {
            v8bf lo = *(const v8bf*)&As[cur][wr + i * 16 + ml][khi];
            v8bf hi = *(const v8bf*)&As[cur][wr + i * 16 + ml][16 + khi];
            af[i] = __builtin_shufflevector(lo, hi, 0, 1, 2, 3, 4, 5, 6, 7,
                                            8, 9, 10, 11, 12, 13, 14, 15);
        }
        #pragma unroll
        for (int j = 0; j < 4; ++j) {
            v8bf lo = *(const v8bf*)&Bs[cur][wc + j * 16 + ml][khi];
            v8bf hi = *(const v8bf*)&Bs[cur][wc + j * 16 + ml][16 + khi];
            bfr[j] = __builtin_shufflevector(lo, hi, 0, 1, 2, 3, 4, 5, 6, 7,
                                             8, 9, 10, 11, 12, 13, 14, 15);
        }
        #pragma unroll
        for (int i = 0; i < 2; ++i)
            #pragma unroll
            for (int j = 0; j < 4; ++j)
                acc[i][j] = __builtin_amdgcn_wmma_f32_16x16x32_bf16(
                    false, af[i], false, bfr[j], (short)0, acc[i][j],
                    false, false);

        // ---- stage step s+1 into the alternate buffer, single barrier -----
        if (more) {
            int nxt = cur ^ 1;
            *(uint4*)&As[nxt][row0][koff]      = a0;
            *(uint4*)&As[nxt][row0 + 64][koff] = a1;
            *(uint4*)&Bs[nxt][row0][koff]      = b0;
            *(uint4*)&Bs[nxt][row0 + 64][koff] = b1;
            __syncthreads();
        }
    }

    // ---- epilogue: D layout (VGPR r -> M = r + 8*(lane>=16), N = lane&15) --
    float* Ct = C + (size_t)t * (size_t)M * Ncol;
    const float* bi = bias ? (bias + (size_t)t * Ncol) : nullptr;
    #pragma unroll
    for (int i = 0; i < 2; ++i) {
        #pragma unroll
        for (int j = 0; j < 4; ++j) {
            int col = bn + wc + j * 16 + ml;
            float bv = bi ? bi[col] : 0.0f;
            #pragma unroll
            for (int r = 0; r < 8; ++r) {
                int row = bm + wr + i * 16 + r + ((lane >> 4) << 3);
                Ct[(size_t)row * Ncol + col] = acc[i][j][r] + bv;
            }
        }
    }
}

// ---------------------------------------------------------------------------
// f32 -> bf16 bulk convert (n multiple of 4)
// ---------------------------------------------------------------------------
__global__ __launch_bounds__(256) void f32_to_bf16(
    const float* __restrict__ src, __bf16* __restrict__ dst, long n)
{
    long i = ((long)blockIdx.x * 256 + threadIdx.x) * 4;
    if (i >= n) return;
    float4 f = *(const float4*)(src + i);
    v4bf o;
    o[0] = (__bf16)f.x; o[1] = (__bf16)f.y; o[2] = (__bf16)f.z; o[3] = (__bf16)f.w;
    *(v4bf*)(dst + i) = o;
}

// Weights [T][K][N] f32 -> [T][N][K] bf16 (K=N=512), 32x32 LDS-tile transpose.
__global__ __launch_bounds__(256) void transpose_to_bf16(
    const float* __restrict__ src, __bf16* __restrict__ dst, int K, int N)
{
    __shared__ float tile[32][33];
    int t = blockIdx.z;
    const float* s = src + (size_t)t * K * N;
    __bf16*      d = dst + (size_t)t * K * N;
    int k0 = blockIdx.y * 32, n0 = blockIdx.x * 32;
    int tx = threadIdx.x & 31, ty = threadIdx.x >> 5;   // 32 x 8
    #pragma unroll
    for (int j = 0; j < 32; j += 8)
        tile[ty + j][tx] = s[(size_t)(k0 + ty + j) * N + n0 + tx];
    __syncthreads();
    #pragma unroll
    for (int j = 0; j < 32; j += 8)
        d[(size_t)(n0 + ty + j) * K + k0 + tx] = (__bf16)tile[tx][ty + j];
}

// ---------------------------------------------------------------------------
// Column statistics: partial sums / sum-of-squares (cols==512, block 256).
// ---------------------------------------------------------------------------
__global__ __launch_bounds__(256) void colstats_partial(
    const float* __restrict__ src, float* __restrict__ sums,
    float* __restrict__ sumsq, int rows, int cols, int chunk)
{
    const int t = blockIdx.y;
    const float* s = src + (size_t)t * rows * cols;
    float* su = sums  + (size_t)t * cols;
    float* sq = sumsq + (size_t)t * cols;
    int r0 = blockIdx.x * chunk;
    int r1 = r0 + chunk; if (r1 > rows) r1 = rows;
    int c = threadIdx.x;
    float s0 = 0.f, q0 = 0.f, s1 = 0.f, q1 = 0.f;
    for (int r = r0; r < r1; ++r) {
        float a = s[(size_t)r * cols + c];
        float b = s[(size_t)r * cols + c + 256];
        s0 += a; q0 += a * a; s1 += b; q1 += b * b;
    }
    atomicAdd(&su[c], s0);       atomicAdd(&sq[c], q0);
    atomicAdd(&su[c + 256], s1); atomicAdd(&sq[c + 256], q1);
}

// Fold BN into per-column scale/shift: y = x*scale + shift
__global__ void finalize_stats(
    const float* __restrict__ sums, const float* __restrict__ sumsq,
    const float* __restrict__ gamma, const float* __restrict__ beta,
    float* __restrict__ scale, float* __restrict__ shift, int rows, int cols)
{
    int t = blockIdx.y;
    int c = blockIdx.x * blockDim.x + threadIdx.x;
    if (c >= cols) return;
    size_t o = (size_t)t * cols + c;
    float mu  = sums[o] / (float)rows;
    float var = sumsq[o] / (float)rows - mu * mu;
    float rs  = rsqrtf(var + 1e-5f);
    float g = gamma[o], b = beta[o];
    scale[o] = rs * g;
    shift[o] = b - mu * rs * g;
}

// v = BN(emb); vn = v/(||v||+1e-8). Writes bf16 copies for the WMMA GEMMs.
__global__ __launch_bounds__(256) void emb_rows(
    const float* __restrict__ emb, const float* __restrict__ scale,
    const float* __restrict__ shift, __bf16* __restrict__ v16,
    __bf16* __restrict__ vn16)
{
    __shared__ float red[256];
    __shared__ float inv;
    int n = blockIdx.x, c = threadIdx.x;
    size_t o = (size_t)n * CD;
    float x0 = emb[o + c]       * scale[c]       + shift[c];
    float x1 = emb[o + c + 256] * scale[c + 256] + shift[c + 256];
    v16[o + c]       = (__bf16)x0;
    v16[o + c + 256] = (__bf16)x1;
    red[c] = x0 * x0 + x1 * x1;
    __syncthreads();
    for (int s = 128; s > 0; s >>= 1) {
        if (c < s) red[c] += red[c + s];
        __syncthreads();
    }
    if (c == 0) inv = 1.0f / (sqrtf(red[0]) + 1e-8f);
    __syncthreads();
    vn16[o + c]       = (__bf16)(x0 * inv);
    vn16[o + c + 256] = (__bf16)(x1 * inv);
}

// Top-30 per row of sim[1024][1024]; one wave32 per row, row in registers.
__global__ __launch_bounds__(32) void topk_rows(
    const float* __restrict__ sim, int* __restrict__ out)
{
    int row  = blockIdx.x;
    int lane = threadIdx.x;
    const float* s = sim + (size_t)row * CN;
    float vals[CN / 32];
    #pragma unroll
    for (int i = 0; i < CN / 32; ++i) vals[i] = s[lane + i * 32];
    unsigned taken = 0u;
    for (int k = 0; k < CK; ++k) {
        float bv = -__builtin_inff();
        int   bi = 0x7fffffff;
        #pragma unroll
        for (int i = 0; i < CN / 32; ++i) {
            if (taken & (1u << i)) continue;
            float vv = vals[i]; int c = lane + i * 32;
            if (vv > bv || (vv == bv && c < bi)) { bv = vv; bi = c; }
        }
        for (int off = 16; off > 0; off >>= 1) {
            float ov = __shfl_xor(bv, off, 32);
            int   oi = __shfl_xor(bi, off, 32);
            if (ov > bv || (ov == bv && oi < bi)) { bv = ov; bi = oi; }
        }
        if (lane == 0) out[(size_t)row * CK + k] = bi;
        if ((bi & 31) == lane) taken |= (1u << (bi >> 5));
    }
}

// In-place BN apply: buf[r] = buf[r]*scale[t] + shift[t]
__global__ __launch_bounds__(256) void normalize_rows(
    float* __restrict__ buf, const float* __restrict__ scale,
    const float* __restrict__ shift, int rowsPerT)
{
    int r = blockIdx.x;
    int t = r / rowsPerT;
    int c = threadIdx.x;
    size_t so  = (size_t)t * CD;
    size_t off = (size_t)r * CD;
    buf[off + c]       = buf[off + c]       * scale[so + c]       + shift[so + c];
    buf[off + c + 256] = buf[off + c + 256] * scale[so + c + 256] + shift[so + c + 256];
}

// hv: normalize 512 distinct rows/type, broadcast-write B=64 copies.
__global__ __launch_bounds__(256) void normalize_hv_tiled(
    const float* __restrict__ raw, const float* __restrict__ scale,
    const float* __restrict__ shift, float* __restrict__ out)
{
    int idx = blockIdx.x;             // t*512 + p
    int t = idx >> 9, p = idx & 511;
    int c = threadIdx.x;
    size_t ro = ((size_t)t * CP + p) * CD;
    size_t so = (size_t)t * CD;
    float v0 = raw[ro + c]       * scale[so + c]       + shift[so + c];
    float v1 = raw[ro + c + 256] * scale[so + c + 256] + shift[so + c + 256];
    for (int b = 0; b < CB; ++b) {
        size_t orow = ((size_t)t * ROWS_PER_T + (size_t)b * CP + p) * CD;
        out[orow + c]       = v0;
        out[orow + c + 256] = v1;
    }
}

// ---------------------------------------------------------------------------
// Workspace layout (bytes); total ~74 MiB
// ---------------------------------------------------------------------------
#define O_EMBSCALE 0u
#define O_EMBSHIFT 2048u
#define O_EMBSUM   4096u
#define O_EMBSQ    6144u
#define O_HXSUM    12288u
#define O_HXSQ     16384u
#define O_HVSUM    20480u
#define O_HVSQ     24576u
#define O_HXSCALE  28672u
#define O_HXSHIFT  32768u
#define O_HVSCALE  36864u
#define O_HVSHIFT  40960u
#define O_ZEROEND  45056u            // memset [0, O_ZEROEND)
#define O_V16      45056u            // 1024*512 bf16 (1 MiB)
#define O_VN16     1093632u          // 1024*512 bf16 (1 MiB)
#define O_SIM      2142208u          // 1024*1024 f32 (4 MiB)
#define O_X16      6336512u          // 64*1024*512 bf16 (64 MiB)
#define O_WX16     73445376u         // 2*512*512 bf16 (1 MiB)
#define O_WV16     74493952u         // 2*512*512 bf16 (1 MiB)
#define O_HV       75542528u         // 2*512*512 f32 (2 MiB)

extern "C" void kernel_launch(void* const* d_in, const int* in_sizes, int n_in,
                              void* d_out, int out_size, void* d_ws, size_t ws_size,
                              hipStream_t stream) {
    (void)in_sizes; (void)n_in; (void)out_size; (void)ws_size;
    const float* x     = (const float*)d_in[0];
    const float* emb   = (const float*)d_in[1];
    const float* emb_g = (const float*)d_in[2];
    const float* emb_b = (const float*)d_in[3];
    const float* Wx    = (const float*)d_in[4];
    const float* bx    = (const float*)d_in[5];
    const float* gx_g  = (const float*)d_in[6];
    const float* gx_b  = (const float*)d_in[7];
    const float* Wv    = (const float*)d_in[8];
    const float* bv    = (const float*)d_in[9];
    const float* gv_g  = (const float*)d_in[10];
    const float* gv_b  = (const float*)d_in[11];

    char*  ws  = (char*)d_ws;
    float* out = (float*)d_out;

    float*  embscale = (float*)(ws + O_EMBSCALE);
    float*  embshift = (float*)(ws + O_EMBSHIFT);
    float*  embsum   = (float*)(ws + O_EMBSUM);
    float*  embsq    = (float*)(ws + O_EMBSQ);
    float*  hxsum    = (float*)(ws + O_HXSUM);
    float*  hxsq     = (float*)(ws + O_HXSQ);
    float*  hvsum    = (float*)(ws + O_HVSUM);
    float*  hvsq     = (float*)(ws + O_HVSQ);
    float*  hxscale  = (float*)(ws + O_HXSCALE);
    float*  hxshift  = (float*)(ws + O_HXSHIFT);
    float*  hvscale  = (float*)(ws + O_HVSCALE);
    float*  hvshift  = (float*)(ws + O_HVSHIFT);
    __bf16* v16      = (__bf16*)(ws + O_V16);
    __bf16* vn16     = (__bf16*)(ws + O_VN16);
    float*  sim      = (float*)(ws + O_SIM);
    __bf16* x16      = (__bf16*)(ws + O_X16);
    __bf16* wx16     = (__bf16*)(ws + O_WX16);
    __bf16* wv16     = (__bf16*)(ws + O_WV16);
    float*  hv_raw   = (float*)(ws + O_HV);

    float* out_hx  = out;                                   // rows [0, 65536)
    float* out_hv  = out + (size_t)HX_ROWS * CD;            // rows [65536, 131072)
    int*   out_idx = (int*)(out + (size_t)2 * HX_ROWS * CD);

    hipMemsetAsync(ws, 0, O_ZEROEND, stream);

    // ---- bf16 operand preparation ----------------------------------------
    const long xN = (long)CB * CN * CL;                     // 33554432
    f32_to_bf16<<<dim3((unsigned)(xN / 4 / 256)), 256, 0, stream>>>(x, x16, xN);
    transpose_to_bf16<<<dim3(CD / 32, CL / 32, CT), 256, 0, stream>>>(Wx, wx16, CL, CD);
    transpose_to_bf16<<<dim3(CD / 32, CD / 32, CT), 256, 0, stream>>>(Wv, wv16, CD, CD);

    // ---- embedding BN -> v16, vn16 ---------------------------------------
    colstats_partial<<<dim3(CN / 128, 1), 256, 0, stream>>>(
        emb, embsum, embsq, CN, CD, 128);
    finalize_stats<<<dim3(CD / 256, 1), 256, 0, stream>>>(
        embsum, embsq, emb_g, emb_b, embscale, embshift, CN, CD);
    emb_rows<<<dim3(CN), 256, 0, stream>>>(emb, embscale, embshift, v16, vn16);

    // ---- sim = vn @ vn^T (WMMA), then top-30 per row ----------------------
    gemm_bf16_wmma<MODE_SIM><<<dim3((CN / 128) * (CN / 128), 1), 256, 0, stream>>>(
        vn16, vn16, nullptr, sim, CN, CN, CD);
    topk_rows<<<dim3(CN), 32, 0, stream>>>(sim, out_idx);

    // ---- hx = BN(x_t @ Wx + bx): GEMM -> d_out, 2-pass BN in place --------
    gemm_bf16_wmma<MODE_HX><<<dim3((ROWS_PER_T / 128) * (CD / 128), CT), 256, 0, stream>>>(
        x16, wx16, bx, out_hx, ROWS_PER_T, CD, CL);
    colstats_partial<<<dim3(ROWS_PER_T / 128, CT), 256, 0, stream>>>(
        out_hx, hxsum, hxsq, ROWS_PER_T, CD, 128);
    finalize_stats<<<dim3(CD / 256, CT), 256, 0, stream>>>(
        hxsum, hxsq, gx_g, gx_b, hxscale, hxshift, ROWS_PER_T, CD);
    normalize_rows<<<dim3(HX_ROWS), 256, 0, stream>>>(
        out_hx, hxscale, hxshift, ROWS_PER_T);

    // ---- hv = BN(v_t @ Wv + bv): 512 distinct rows/type, then tile --------
    gemm_bf16_wmma<MODE_HV><<<dim3((CP / 128) * (CD / 128), CT), 256, 0, stream>>>(
        v16, wv16, bv, hv_raw, CP, CD, CD);
    colstats_partial<<<dim3(CP / 128, CT), 256, 0, stream>>>(
        hv_raw, hvsum, hvsq, CP, CD, 128);
    finalize_stats<<<dim3(CD / 256, CT), 256, 0, stream>>>(
        hvsum, hvsq, gv_g, gv_b, hvscale, hvshift, CP, CD);
    normalize_hv_tiled<<<dim3(CT * CP), 256, 0, stream>>>(
        hv_raw, hvscale, hvshift, out_hv);
}
